// ARWaveNO_61065845015069
// MI455X (gfx1250) — matrix operand order repaired and checked by reading, not compiled, not run on hardware
//
#include <hip/hip_runtime.h>
#include <hip/hip_bf16.h>

typedef __attribute__((ext_vector_type(16))) _Float16 v16h;
typedef __attribute__((ext_vector_type(8)))  _Float16 v8h;
typedef __attribute__((ext_vector_type(8)))  float    v8f;

#define NB      8
#define NXC     128
#define KQ      8
#define HD      256
#define NHEADS  8
#define DHEAD   32
#define NLAY    4
#define FFND    1024
#define QTOK    1024           // KQ*NXC
#define NTOK    1152           // NXC + QTOK
#define MROWS   (NB*NTOK)      // 9216

static __device__ __forceinline__ v16h cat8(v8h lo, v8h hi) {
  return __builtin_shufflevector(lo, hi, 0,1,2,3,4,5,6,7,8,9,10,11,12,13,14,15);
}
static __device__ __forceinline__ v8f wmma16(v16h a, v16h b, v8f c) {
  return __builtin_amdgcn_wmma_f32_16x16x32_f16(false, a, false, b, (short)0, c, false, false);
}
// Map packed row m -> interleaved token row (per batch: NXC cell rows then QTOK query rows)
static __device__ __forceinline__ long maprow(int m, int seg, int off) {
  return seg ? ((long)(m / seg) * NTOK + off + (m % seg)) : (long)m;
}

// ---------------------------------------------------------------------------
// Generic f16 WMMA GEMM: out = A(f16,MxK) @ W^T(f16,[N][K]) + bias [;relu][;+res]
// 64x64 tile per 128-thread block (4 waves, each a 16x64 strip).
// B (weights) arrive pre-transposed [N][K] so BOTH tiles stage as b128 copies.
// Register-pipelined: next K-tile global loads issued before the WMMAs.
// ---------------------------------------------------------------------------
__global__ __launch_bounds__(128) void gemm_f16(
    const _Float16* __restrict__ A, int lda, int a_seg, int a_off,
    const _Float16* __restrict__ Wt,           // [N][Ktot] transposed weights
    const float* __restrict__ bias,
    float* __restrict__ outF, _Float16* __restrict__ outH, int ldo,
    int o_seg, int o_off, int outh_mode,       // 0: row-major; 1: [b,head,dh,token]
    const float* __restrict__ residual, int do_relu, int Ktot)
{
  __shared__ alignas(32) _Float16 At[64][32];
  __shared__ alignas(32) _Float16 Bt[64][32];   // [n][k] tile
  const int tid  = threadIdx.x;
  const int wave = tid >> 5, lane = tid & 31, ll = lane & 15, lh = lane >> 4;
  const int mbase = blockIdx.x * 64, nbase = blockIdx.y * 64;

  v8f acc0 = {}, acc1 = {}, acc2 = {}, acc3 = {};

  const int rr = tid >> 1, cc = (tid & 1) * 16;           // coop-copy coords (both tiles)
  const long arow = maprow(mbase + rr, a_seg, a_off);
  const _Float16* aptr = A  + (size_t)arow * lda + cc;
  const _Float16* bptr = Wt + (size_t)(nbase + rr) * Ktot + cc;

  v8h a0 = *(const v8h*)(aptr);
  v8h a1 = *(const v8h*)(aptr + 8);
  v8h b0 = *(const v8h*)(bptr);
  v8h b1 = *(const v8h*)(bptr + 8);

  for (int kk = 0; kk < Ktot; kk += 32) {
    __syncthreads();                       // previous compute done; safe to overwrite LDS
    *(v8h*)&At[rr][cc]     = a0;
    *(v8h*)&At[rr][cc + 8] = a1;
    *(v8h*)&Bt[rr][cc]     = b0;
    *(v8h*)&Bt[rr][cc + 8] = b1;
    __syncthreads();
    if (kk + 32 < Ktot) {                  // preload next K-tile; overlaps WMMAs below
      a0 = *(const v8h*)(aptr + kk + 32);
      a1 = *(const v8h*)(aptr + kk + 40);
      b0 = *(const v8h*)(bptr + kk + 32);
      b1 = *(const v8h*)(bptr + kk + 40);
      if (kk + 64 < Ktot) {                // gfx1250 global_prefetch two tiles ahead
        __builtin_prefetch(aptr + kk + 64, 0, 1);
        __builtin_prefetch(bptr + kk + 64, 0, 1);
      }
    }
    // A fragment (16x32): lane = M row, 16 contiguous K at 16*lh
    const _Float16* ar = &At[wave * 16 + ll][0];
    v16h af = *(const v16h*)(ar + 16 * lh);
    // B fragments (32x16): lane = N col, 16 contiguous K halves from [n][k] tile
    v16h bf0 = *(const v16h*)&Bt[ 0 + ll][16 * lh];
    v16h bf1 = *(const v16h*)&Bt[16 + ll][16 * lh];
    v16h bf2 = *(const v16h*)&Bt[32 + ll][16 * lh];
    v16h bf3 = *(const v16h*)&Bt[48 + ll][16 * lh];
    acc0 = wmma16(af, bf0, acc0);
    acc1 = wmma16(af, bf1, acc1);
    acc2 = wmma16(af, bf2, acc2);
    acc3 = wmma16(af, bf3, acc3);
  }

  v8f accs[4] = {acc0, acc1, acc2, acc3};
#pragma unroll
  for (int j = 0; j < 4; ++j) {
    int col = nbase + j * 16 + ll;
    float bv = bias ? bias[col] : 0.f;
#pragma unroll
    for (int r = 0; r < 8; ++r) {
      int m = mbase + wave * 16 + r + 8 * lh;       // C layout: row = r + 8*(lane>>4)
      long orow = maprow(m, o_seg, o_off);
      float v = accs[j][r] + bv;
      if (do_relu) v = fmaxf(v, 0.f);
      if (residual) v += residual[(size_t)orow * HD + col];
      if (outF) outF[(size_t)orow * ldo + col] = v;
      if (outH) {
        if (outh_mode == 0) {
          outH[(size_t)orow * ldo + col] = (_Float16)v;
        } else {            // V for attention: [b][head][dh][token]
          int bidx = (int)(orow / NTOK), n = (int)(orow % NTOK);
          int head = col >> 5, d = col & 31;
          outH[(((size_t)bidx * NHEADS + head) * DHEAD + d) * NTOK + n] = (_Float16)v;
        }
      }
    }
  }
}

// ---------------------------------------------------------------------------
// Flash attention: one wave = one 16-row query tile; S=QK^T and P*V via WMMA,
// online softmax in f32. DH=32 -> one WMMA per 16x16 score tile.
// V arrives pre-transposed [b][head][dh][token] -> contiguous b128 staging.
// ---------------------------------------------------------------------------
__global__ __launch_bounds__(128) void attn_flash(
    const _Float16* __restrict__ qb, const _Float16* __restrict__ kb,
    const _Float16* __restrict__ vtg, const float* __restrict__ cbias,
    _Float16* __restrict__ ctx16)
{
  __shared__ alignas(32) _Float16 Kt[32][32];      // [kv][dh]  (== B^T for S)
  __shared__ alignas(32) _Float16 Vt[32][32];      // [dh][kv]  (== B^T for P@V)
  __shared__ alignas(32) _Float16 Pw[4][16][32];   // per-wave P tile
  const int tid  = threadIdx.x;
  const int wave = tid >> 5, lane = tid & 31, ll = lane & 15, lh = lane >> 4;
  const int b = blockIdx.y >> 3, head = blockIdx.y & 7;
  const int qtile = blockIdx.x * 64 + wave * 16;
  const size_t hb = (size_t)head * DHEAD;
  const _Float16* vbase = vtg + ((size_t)b * NHEADS + head) * DHEAD * NTOK;

  // Q fragment, reused over all kv chunks
  const _Float16* qrow = qb + ((size_t)(b * NTOK + qtile + ll)) * HD + hb;
  v16h aq = *(const v16h*)(qrow + 16 * lh);

  float mrun[8], lrun[8];
  v8f cx0 = {}, cx1 = {};
#pragma unroll
  for (int r = 0; r < 8; ++r) { mrun[r] = -1e30f; lrun[r] = 0.f; }
  const float scale = 0.17677669529663687f;  // 1/sqrt(32)

  for (int kvb = 0; kvb < NTOK; kvb += 32) {
    __syncthreads();
    {   // cooperative stage: K chunk row-major, V chunk already transposed in global
      int kv = tid >> 2, d0 = (tid & 3) * 8;
      v8h kk = *(const v8h*)(kb + ((size_t)(b * NTOK + kvb + kv)) * HD + hb + d0);
      *(v8h*)&Kt[kv][d0] = kk;
      int d = tid >> 2, kv0 = (tid & 3) * 8;
      v8h vv = *(const v8h*)(vbase + (size_t)d * NTOK + kvb + kv0);
      *(v8h*)&Vt[d][kv0] = vv;
    }
    __syncthreads();

    v16h bk0 = *(const v16h*)&Kt[ 0 + ll][16 * lh];
    v16h bk1 = *(const v16h*)&Kt[16 + ll][16 * lh];
    v8f z = {};
    v8f s0 = wmma16(aq, bk0, z);
    v8f s1 = wmma16(aq, bk1, z);

    float p0[8], p1[8];
#pragma unroll
    for (int r = 0; r < 8; ++r) {
      int qg  = qtile + r + 8 * lh;
      int kg0 = kvb + ll, kg1 = kvb + 16 + ll;
      float a = s0[r] * scale, c = s1[r] * scale;
      if (qg >= NXC) {   // mask[:, NX:, :NX] = char_bias
        const float* cb = cbias + ((size_t)b * QTOK + (qg - NXC)) * NXC;
        if (kg0 < NXC) a += cb[kg0];
        if (kg1 < NXC) c += cb[kg1];
      }
      float rm = fmaxf(a, c);
#pragma unroll
      for (int off = 1; off < 16; off <<= 1) rm = fmaxf(rm, __shfl_xor(rm, off, 32));
      float mn = fmaxf(mrun[r], rm);
      float al = __expf(mrun[r] - mn);
      float e0 = __expf(a - mn), e1 = __expf(c - mn);
      float rs = e0 + e1;
#pragma unroll
      for (int off = 1; off < 16; off <<= 1) rs += __shfl_xor(rs, off, 32);
      lrun[r] = lrun[r] * al + rs;
      mrun[r] = mn;
      cx0[r] *= al; cx1[r] *= al;
      p0[r] = e0; p1[r] = e1;
    }
    // store P (f16) to per-wave LDS, regather in A-operand layout
#pragma unroll
    for (int r = 0; r < 8; ++r) {
      Pw[wave][r + 8 * lh][ll]      = (_Float16)p0[r];
      Pw[wave][r + 8 * lh][16 + ll] = (_Float16)p1[r];
    }
    asm volatile("s_wait_dscnt 0" ::: "memory");   // intra-wave LDS RAW
    const _Float16* pr = &Pw[wave][ll][0];
    v16h ap = *(const v16h*)(pr + 16 * lh);
    v16h bv0 = *(const v16h*)&Vt[ 0 + ll][16 * lh];
    v16h bv1 = *(const v16h*)&Vt[16 + ll][16 * lh];
    cx0 = wmma16(ap, bv0, cx0);
    cx1 = wmma16(ap, bv1, cx1);
  }

#pragma unroll
  for (int r = 0; r < 8; ++r) {
    int qg = qtile + r + 8 * lh;
    float inv = 1.f / lrun[r];
    size_t dst = ((size_t)(b * NTOK + qg)) * HD + hb;
    ctx16[dst + ll]      = (_Float16)(cx0[r] * inv);
    ctx16[dst + 16 + ll] = (_Float16)(cx1[r] * inv);
  }
}

// ---------------------------------------------------------------------------
// LayerNorm over H=256 (one block per row), dual f32/f16 output
// ---------------------------------------------------------------------------
__global__ __launch_bounds__(256) void layernorm(
    const float* __restrict__ in, const float* __restrict__ g,
    const float* __restrict__ bt, float* __restrict__ xo, _Float16* __restrict__ xh)
{
  __shared__ float red[256];
  int row = blockIdx.x, tid = threadIdx.x;
  float v = in[(size_t)row * HD + tid];
  red[tid] = v; __syncthreads();
  for (int s = 128; s > 0; s >>= 1) { if (tid < s) red[tid] += red[tid + s]; __syncthreads(); }
  float mean = red[0] / HD; __syncthreads();
  float d = v - mean;
  red[tid] = d * d; __syncthreads();
  for (int s = 128; s > 0; s >>= 1) { if (tid < s) red[tid] += red[tid + s]; __syncthreads(); }
  float var = red[0] / HD;
  float y = d * rsqrtf(var + 1e-5f) * g[tid] + bt[tid];
  xo[(size_t)row * HD + tid] = y;
  xh[(size_t)row * HD + tid] = (_Float16)y;
}

// ---------------------------------------------------------------------------
// Encoders: features, 34->256 MLP layer 1, weight convert+transpose, head
// ---------------------------------------------------------------------------
__global__ __launch_bounds__(128) void cell_features(
    const float* __restrict__ sr, const float* __restrict__ dx, float* __restrict__ feat)
{
  __shared__ float sc[128];
  int b = blockIdx.x, tid = threadIdx.x;
  float ks = sr[b * NXC + tid];
  sc[tid] = ks; __syncthreads();
  for (int off = 1; off < 128; off <<= 1) {   // inclusive scan
    float u = (tid >= off) ? sc[tid - off] : 0.f;
    __syncthreads();
    sc[tid] += u;
    __syncthreads();
  }
  float mass = sc[tid] * dx[b];
  float* f = feat + ((size_t)b * NXC + tid) * 34;
  f[0] = ks; f[17] = mass;
  float fr = 3.14159265358979323846f;
#pragma unroll
  for (int j = 0; j < 8; ++j) {
    f[1 + j]  = sinf(ks * fr);   f[9 + j]  = cosf(ks * fr);
    f[18 + j] = sinf(mass * fr); f[26 + j] = cosf(mass * fr);
    fr *= 2.f;
  }
}

__global__ __launch_bounds__(256) void query_features(
    const float* __restrict__ t, const float* __restrict__ x, float* __restrict__ feat)
{
  int m = blockIdx.x * 256 + threadIdx.x;          // 0..8191 = b*QTOK + q
  float tv = t[m], xv = x[m];
  float* f = feat + (size_t)m * 34;
  f[0] = tv; f[17] = xv;
  float fr = 3.14159265358979323846f;
#pragma unroll
  for (int j = 0; j < 8; ++j) {
    f[1 + j]  = sinf(tv * fr); f[9 + j]  = cosf(tv * fr);
    f[18 + j] = sinf(xv * fr); f[26 + j] = cosf(xv * fr);
    fr *= 2.f;
  }
}

__global__ __launch_bounds__(256) void mlp_layer1(
    const float* __restrict__ feat, const float* __restrict__ w1,
    const float* __restrict__ b1, _Float16* __restrict__ oh)
{
  __shared__ float f[34];
  int row = blockIdx.x, tid = threadIdx.x;
  if (tid < 34) f[tid] = feat[(size_t)row * 34 + tid];
  __syncthreads();
  float s = b1[tid];
#pragma unroll
  for (int i = 0; i < 34; ++i) s += f[i] * w1[i * HD + tid];
  oh[(size_t)row * HD + tid] = (_Float16)fmaxf(s, 0.f);
}

// f32 [K][N] weight -> f16 transposed [N][K] (one-time; off the GEMM hot loop)
__global__ void conv_transpose_w(const float* __restrict__ s, _Float16* __restrict__ d,
                                 int K, int N)
{
  int i = blockIdx.x * 256 + threadIdx.x;
  if (i < K * N) {
    int k = i / N, n = i % N;
    d[(size_t)n * K + k] = (_Float16)s[i];
  }
}

__global__ __launch_bounds__(256) void head_out(
    const _Float16* __restrict__ h, const float* __restrict__ w2,
    const float* __restrict__ b2, float* __restrict__ out)
{
  int m = blockIdx.x * 256 + threadIdx.x;          // output layout is identity
  float s = b2[0];
  for (int i = 0; i < HD; ++i) s += (float)h[(size_t)m * HD + i] * w2[i];
  out[m] = fminf(fmaxf(s, 0.f), 1.f);
}

// ---------------------------------------------------------------------------
extern "C" void kernel_launch(void* const* d_in, const int* in_sizes, int n_in,
                              void* d_out, int out_size, void* d_ws, size_t ws_size,
                              hipStream_t stream)
{
  (void)in_sizes; (void)n_in; (void)out_size; (void)ws_size;
  const float* state_row = (const float*)d_in[0];
  const float* t_coords  = (const float*)d_in[1];
  const float* x_coords  = (const float*)d_in[2];
  const float* dx        = (const float*)d_in[3];
  const float* char_bias = (const float*)d_in[4];
  const float* enc_w1 = (const float*)d_in[5];
  const float* enc_b1 = (const float*)d_in[6];
  const float* enc_w2 = (const float*)d_in[7];
  const float* enc_b2 = (const float*)d_in[8];
  const float* q_w1   = (const float*)d_in[9];
  const float* q_b1   = (const float*)d_in[10];
  const float* q_w2   = (const float*)d_in[11];
  const float* q_b2   = (const float*)d_in[12];
  const int L0 = 13;   // per layer: wq,bq,wk,bk,wv,bv,wo,bo,ln1_g,ln1_b,ff_w1,ff_b1,ff_w2,ff_b2,ln2_g,ln2_b
  const float* out_w1 = (const float*)d_in[L0 + 16 * NLAY + 0];
  const float* out_b1 = (const float*)d_in[L0 + 16 * NLAY + 1];
  const float* out_w2 = (const float*)d_in[L0 + 16 * NLAY + 2];
  const float* out_b2 = (const float*)d_in[L0 + 16 * NLAY + 3];

  // ---- workspace carve ----
  size_t off = 0;
  auto alloc = [&](size_t bytes) -> void* {
    void* p = (char*)d_ws + off;
    off = (off + bytes + 255) & ~(size_t)255;
    return p;
  };
  float*     xf32  = (float*)alloc((size_t)MROWS * HD * 4);
  float*     tmpf  = (float*)alloc((size_t)MROWS * HD * 4);
  _Float16*  x16   = (_Float16*)alloc((size_t)MROWS * HD * 2);
  _Float16*  q16   = (_Float16*)alloc((size_t)MROWS * HD * 2);
  _Float16*  k16   = (_Float16*)alloc((size_t)MROWS * HD * 2);
  _Float16*  vT16  = (_Float16*)alloc((size_t)MROWS * HD * 2);  // [b][head][dh][token]
  _Float16*  c16   = (_Float16*)alloc((size_t)MROWS * HD * 2);
  _Float16*  hff   = (_Float16*)alloc((size_t)MROWS * FFND * 2);
  _Float16*  hc    = (_Float16*)alloc((size_t)NB * NXC * HD * 2);
  _Float16*  hq    = (_Float16*)alloc((size_t)NB * QTOK * HD * 2);
  _Float16*  hfin  = (_Float16*)alloc((size_t)NB * QTOK * HD * 2);
  float*     fc    = (float*)alloc((size_t)NB * NXC * 34 * 4);
  float*     fq    = (float*)alloc((size_t)NB * QTOK * 34 * 4);
  _Float16*  encw2h = (_Float16*)alloc((size_t)HD * HD * 2);
  _Float16*  qw2h   = (_Float16*)alloc((size_t)HD * HD * 2);
  _Float16*  outw1h = (_Float16*)alloc((size_t)HD * HD * 2);
  _Float16 *wqh[NLAY], *wkh[NLAY], *wvh[NLAY], *woh[NLAY], *ff1h[NLAY], *ff2h[NLAY];
  for (int l = 0; l < NLAY; ++l) {
    wqh[l]  = (_Float16*)alloc((size_t)HD * HD * 2);
    wkh[l]  = (_Float16*)alloc((size_t)HD * HD * 2);
    wvh[l]  = (_Float16*)alloc((size_t)HD * HD * 2);
    woh[l]  = (_Float16*)alloc((size_t)HD * HD * 2);
    ff1h[l] = (_Float16*)alloc((size_t)HD * FFND * 2);
    ff2h[l] = (_Float16*)alloc((size_t)FFND * HD * 2);
  }

  auto convT = [&](const float* s, _Float16* d, int K, int N) {
    conv_transpose_w<<<(K * N + 255) / 256, 256, 0, stream>>>(s, d, K, N);
  };
  convT(enc_w2, encw2h, HD, HD);
  convT(q_w2,   qw2h,   HD, HD);
  convT(out_w1, outw1h, HD, HD);
  for (int l = 0; l < NLAY; ++l) {
    const int base = L0 + 16 * l;
    convT((const float*)d_in[base + 0],  wqh[l],  HD, HD);
    convT((const float*)d_in[base + 2],  wkh[l],  HD, HD);
    convT((const float*)d_in[base + 4],  wvh[l],  HD, HD);
    convT((const float*)d_in[base + 6],  woh[l],  HD, HD);
    convT((const float*)d_in[base + 10], ff1h[l], HD, FFND);
    convT((const float*)d_in[base + 12], ff2h[l], FFND, HD);
  }

  auto gemm = [&](const _Float16* A, int lda, int a_seg, int a_off,
                  const _Float16* Wt, const float* bias,
                  float* oF, _Float16* oH, int ldo, int o_seg, int o_off, int ohm,
                  const float* res, int relu, int M, int Nn, int Kt) {
    dim3 g(M / 64, Nn / 64);
    gemm_f16<<<g, 128, 0, stream>>>(A, lda, a_seg, a_off, Wt, bias,
                                    oF, oH, ldo, o_seg, o_off, ohm, res, relu, Kt);
  };

  // ---- token embedding ----
  cell_features<<<NB, 128, 0, stream>>>(state_row, dx, fc);
  query_features<<<(NB * QTOK) / 256, 256, 0, stream>>>(t_coords, x_coords, fq);
  mlp_layer1<<<NB * NXC, 256, 0, stream>>>(fc, enc_w1, enc_b1, hc);
  mlp_layer1<<<NB * QTOK, 256, 0, stream>>>(fq, q_w1, q_b1, hq);
  // second encoder layers via WMMA; scatter into interleaved token tensor
  gemm(hc, HD, 0, 0, encw2h, enc_b2, xf32, x16, HD, NXC, 0,    0, nullptr, 0, NB * NXC,  HD, HD);
  gemm(hq, HD, 0, 0, qw2h,   q_b2,   xf32, x16, HD, QTOK, NXC, 0, nullptr, 0, NB * QTOK, HD, HD);

  // ---- transformer layers ----
  for (int l = 0; l < NLAY; ++l) {
    const int base = L0 + 16 * l;
    const float* bq   = (const float*)d_in[base + 1];
    const float* bk   = (const float*)d_in[base + 3];
    const float* bv   = (const float*)d_in[base + 5];
    const float* bo   = (const float*)d_in[base + 7];
    const float* ln1g = (const float*)d_in[base + 8];
    const float* ln1b = (const float*)d_in[base + 9];
    const float* fb1  = (const float*)d_in[base + 11];
    const float* fb2  = (const float*)d_in[base + 13];
    const float* ln2g = (const float*)d_in[base + 14];
    const float* ln2b = (const float*)d_in[base + 15];

    gemm(x16, HD, 0, 0, wqh[l], bq, nullptr, q16,  HD, 0, 0, 0, nullptr, 0, MROWS, HD, HD);
    gemm(x16, HD, 0, 0, wkh[l], bk, nullptr, k16,  HD, 0, 0, 0, nullptr, 0, MROWS, HD, HD);
    gemm(x16, HD, 0, 0, wvh[l], bv, nullptr, vT16, HD, 0, 0, 1, nullptr, 0, MROWS, HD, HD);
    attn_flash<<<dim3(NTOK / 64, NB * NHEADS), 128, 0, stream>>>(q16, k16, vT16, char_bias, c16);
    gemm(c16, HD, 0, 0, woh[l], bo, tmpf, nullptr, HD, 0, 0, 0, xf32, 0, MROWS, HD, HD);
    layernorm<<<MROWS, 256, 0, stream>>>(tmpf, ln1g, ln1b, xf32, x16);
    gemm(x16, HD,   0, 0, ff1h[l], fb1, nullptr, hff, FFND, 0, 0, 0, nullptr, 1, MROWS, FFND, HD);
    gemm(hff, FFND, 0, 0, ff2h[l], fb2, tmpf, nullptr, HD,  0, 0, 0, xf32,    0, MROWS, HD, FFND);
    layernorm<<<MROWS, 256, 0, stream>>>(tmpf, ln2g, ln2b, xf32, x16);
  }

  // ---- output head (query tokens only) ----
  gemm(x16, HD, QTOK, NXC, outw1h, out_b1, nullptr, hfin, HD, 0, 0, 0, nullptr, 1, NB * QTOK, HD, HD);
  head_out<<<(NB * QTOK) / 256, 256, 0, stream>>>(hfin, out_w2, out_b2, (float*)d_out);
}
